// AdapterFormer_74225624809658
// MI455X (gfx1250) — compile-verified, matrix operand order checked by reading, not run on hardware
//
#include <hip/hip_runtime.h>
#include <cstdint>
#include <cstddef>

// ---------------------------------------------------------------------------
// AdapterFormer block for MI455X (gfx1250, wave32, WMMA 16x16x32 f16->f32)
// L=197, BT=64, D=768, H=12, HD=64, E=4 (5 branches), DF=3072
// GEMM tile staging uses CDNA5 async global->LDS copies (ASYNCcnt).
// ---------------------------------------------------------------------------

typedef _Float16 h8   __attribute__((ext_vector_type(8)));
typedef _Float16 v16h __attribute__((ext_vector_type(16)));
typedef float    v8f  __attribute__((ext_vector_type(8)));

static constexpr int LSEQ = 197;
static constexpr int BTCH = 64;
static constexpr int DMOD = 768;
static constexpr int NHEAD = 12;
static constexpr int HDIM = 64;
static constexpr int TDIM = 2304;   // 3*D
static constexpr int DFF  = 3072;
static constexpr int NTOK = LSEQ * BTCH;   // 12608 = 197*64

__device__ __forceinline__ v8f wmma32(v16h a, v16h b, v8f c) {
  // v_wmma_f32_16x16x32_f16
  return __builtin_amdgcn_wmma_f32_16x16x32_f16(false, a, false, b,
                                                (short)0, c, false, false);
}

__device__ __forceinline__ v16h pack16(h8 lo, h8 hi) {
  v16h r;
#pragma unroll
  for (int i = 0; i < 8; ++i) { r[i] = lo[i]; r[i + 8] = hi[i]; }
  return r;
}

__device__ __forceinline__ v16h zero16() {
  v16h r;
#pragma unroll
  for (int i = 0; i < 16; ++i) r[i] = (_Float16)0.f;
  return r;
}

// A/B fragment gather (ISA 7.12.2, 16-bit A 16x32):
// lane L (0..15): rows M=L, K = 0..7 and 16..23
// lane L (16..31): rows M=L-16, K = 8..15 and 24..31
__device__ __forceinline__ v16h frag_ld(const _Float16* base, int row0, int stride) {
  const int lane = threadIdx.x & 31;
  const _Float16* p = base + (size_t)(row0 + (lane & 15)) * stride + ((lane >> 4) << 3);
  return pack16(*(const h8*)p, *(const h8*)(p + 16));
}

// CDNA5 async global->LDS copy, 16B per lane (ASYNCcnt tracked).
// lds_off: byte offset within workgroup LDS (low 32 bits of flat shared addr).
__device__ __forceinline__ void async_copy_b128(uint32_t lds_off, uint64_t gaddr) {
  asm volatile("global_load_async_to_lds_b128 %0, %1, off"
               :: "v"(lds_off), "v"(gaddr) : "memory");
}
__device__ __forceinline__ void wait_async0() {
  asm volatile("s_wait_asynccnt 0x0" ::: "memory");
}
__device__ __forceinline__ uint32_t lds_off_of(const void* p) {
  return (uint32_t)(uintptr_t)p;
}

// ---------------------------------------------------------------------------
// f32 -> f16 conversion (weights)
// ---------------------------------------------------------------------------
__global__ void cvt_kernel(const float* __restrict__ in, _Float16* __restrict__ out,
                           size_t n) {
  for (size_t i = (size_t)blockIdx.x * blockDim.x + threadIdx.x; i < n;
       i += (size_t)gridDim.x * blockDim.x)
    out[i] = (_Float16)in[i];
}

// ---------------------------------------------------------------------------
// LayerNorm over rows of 768, wave-per-row, writes f16
// ---------------------------------------------------------------------------
__global__ __launch_bounds__(256) void ln_kernel(const float* __restrict__ X,
                                                 const float* __restrict__ g,
                                                 const float* __restrict__ bt,
                                                 _Float16* __restrict__ out,
                                                 int nrows) {
  const int row = blockIdx.x * 8 + (threadIdx.x >> 5);
  if (row >= nrows) return;
  const int lane = threadIdx.x & 31;
  const float* x = X + (size_t)row * DMOD;
  float v[24];
  float s = 0.f;
#pragma unroll
  for (int i = 0; i < 24; ++i) { v[i] = x[lane + 32 * i]; s += v[i]; }
#pragma unroll
  for (int d = 1; d < 32; d <<= 1) s += __shfl_xor(s, d, 32);
  const float mu = s * (1.f / DMOD);
  float var = 0.f;
#pragma unroll
  for (int i = 0; i < 24; ++i) { float t = v[i] - mu; var += t * t; }
#pragma unroll
  for (int d = 1; d < 32; d <<= 1) var += __shfl_xor(var, d, 32);
  const float inv = rsqrtf(var * (1.f / DMOD) + 1e-5f);
  _Float16* o = out + (size_t)row * DMOD;
#pragma unroll
  for (int i = 0; i < 24; ++i) {
    const int c = lane + 32 * i;
    o[c] = (_Float16)((v[i] - mu) * inv * g[c] + bt[c]);
  }
}

// ---------------------------------------------------------------------------
// Router: softmax(A @ W^T + b), 5 outputs per row, wave-per-row
// ---------------------------------------------------------------------------
__global__ __launch_bounds__(256) void router_kernel(const _Float16* __restrict__ A,
                                                     int K,
                                                     const float* __restrict__ W,
                                                     const float* __restrict__ bias,
                                                     float* __restrict__ out,
                                                     int nrows) {
  const int row = blockIdx.x * 8 + (threadIdx.x >> 5);
  if (row >= nrows) return;
  const int lane = threadIdx.x & 31;
  float acc[5] = {0.f, 0.f, 0.f, 0.f, 0.f};
  const _Float16* a = A + (size_t)row * K;
  for (int c = lane; c < K; c += 32) {
    const float av = (float)a[c];
#pragma unroll
    for (int e = 0; e < 5; ++e) acc[e] += av * W[(size_t)e * K + c];
  }
#pragma unroll
  for (int e = 0; e < 5; ++e) {
#pragma unroll
    for (int d = 1; d < 32; d <<= 1) acc[e] += __shfl_xor(acc[e], d, 32);
  }
  if (lane == 0) {
    float m = -1e30f;
#pragma unroll
    for (int e = 0; e < 5; ++e) { acc[e] += bias[e]; m = fmaxf(m, acc[e]); }
    float s = 0.f;
#pragma unroll
    for (int e = 0; e < 5; ++e) { acc[e] = __expf(acc[e] - m); s += acc[e]; }
    const float inv = 1.f / s;
#pragma unroll
    for (int e = 0; e < 5; ++e) out[(size_t)row * 5 + e] = acc[e] * inv;
  }
}

// ---------------------------------------------------------------------------
// Generic NT WMMA GEMM: out[n,m] = combine_e( A[n,:] . W_e[m,:] )
// A: [M=12608, K] f16, W_e: [NC, K] f16.  Block tile 64(M) x 128(N) x 32(K).
// 8 waves: wave = (wm, wn) in 2 x 4 grid, each computes 32x32 (2x2 WMMA tiles).
// Tiles staged with async global->LDS copies (no VGPR round-trip).
// NB=1: plain bias epilogue.  NB=5: router-weighted sum of 5 branches.
// ---------------------------------------------------------------------------
template <int NB>
__global__ __launch_bounds__(256) void gemm_wmma(
    const _Float16* __restrict__ A, int K, int NC,
    const _Float16* __restrict__ W0, const _Float16* __restrict__ WE,
    const float* __restrict__ bias0, const float* __restrict__ biasE,
    const float* __restrict__ R,       // [M,5] routing weights (NB==5)
    const float* __restrict__ resid,   // optional f32 residual [M,NC]
    float* __restrict__ outF,          // optional f32 output
    _Float16* __restrict__ outH,       // optional f16 output
    int act) {                         // 1 = quickGELU
  __shared__ _Float16 sA[64 * 32];
  __shared__ _Float16 sW[NB * 128 * 32];
  const int tid = threadIdx.x, lane = tid & 31, wave = tid >> 5;
  const int wm = wave & 1, wn = wave >> 1;
  const int m0 = blockIdx.x * 64, n0 = blockIdx.y * 128;

  v8f acc[NB][2][2];
#pragma unroll
  for (int e = 0; e < NB; ++e)
#pragma unroll
    for (int mt = 0; mt < 2; ++mt)
#pragma unroll
      for (int nt = 0; nt < 2; ++nt)
#pragma unroll
        for (int r = 0; r < 8; ++r) acc[e][mt][nt][r] = 0.f;

  // per-thread staging coordinates (constant across k-steps)
  const int arow = tid >> 2, acol = (tid & 3) << 3;     // A: 16B / thread
  const int wrow = tid >> 1, wcol = (tid & 1) << 4;     // W: 32B / thread / branch
  const uint32_t aoff = lds_off_of(&sA[arow * 32 + acol]);

  const int nK = K >> 5;
  for (int kk = 0; kk < nK; ++kk) {
    __syncthreads();  // all waves done reading previous tiles
    async_copy_b128(aoff,
                    (uint64_t)&A[(size_t)(m0 + arow) * K + (kk << 5) + acol]);
#pragma unroll
    for (int e = 0; e < NB; ++e) {
      const _Float16* W = (e == 0) ? W0 : WE + (size_t)(e - 1) * NC * K;
      const _Float16* gp = &W[(size_t)(n0 + wrow) * K + (kk << 5) + wcol];
      const uint32_t woff = lds_off_of(&sW[e * 4096 + wrow * 32 + wcol]);
      async_copy_b128(woff, (uint64_t)gp);
      async_copy_b128(woff + 16, (uint64_t)(gp + 8));
    }
    wait_async0();    // this wave's async copies landed in LDS
    __syncthreads();  // everyone's copies landed

    v16h aF[2];
    aF[0] = frag_ld(sA, wm * 32, 32);
    aF[1] = frag_ld(sA, wm * 32 + 16, 32);
#pragma unroll
    for (int e = 0; e < NB; ++e) {
#pragma unroll
      for (int nt = 0; nt < 2; ++nt) {
        v16h wF = frag_ld(&sW[e * 4096], wn * 32 + nt * 16, 32);
        acc[e][0][nt] = wmma32(aF[0], wF, acc[e][0][nt]);
        acc[e][1][nt] = wmma32(aF[1], wF, acc[e][1][nt]);
      }
    }
  }

  // Epilogue. C layout: lane -> n = lane&15, VGPR r -> m = r + 8*(lane>=16)
  const int nl = lane & 15, mh = (lane >> 4) << 3;
#pragma unroll
  for (int mt = 0; mt < 2; ++mt) {
#pragma unroll
    for (int nt = 0; nt < 2; ++nt) {
      const int col = n0 + wn * 32 + nt * 16 + nl;
#pragma unroll
      for (int r = 0; r < 8; ++r) {
        const int row = m0 + wm * 32 + mt * 16 + mh + r;
        float v;
        if (NB == 1) {
          v = acc[0][mt][nt][r] + (bias0 ? bias0[col] : 0.f);
        } else {
          v = 0.f;
#pragma unroll
          for (int e = 0; e < NB; ++e) {
            const float bb =
                (e == 0) ? bias0[col] : biasE[(size_t)(e - 1) * NC + col];
            v += R[(size_t)row * 5 + e] * (acc[e][mt][nt][r] + bb);
          }
        }
        if (resid) v += resid[(size_t)row * NC + col];
        if (act) v = v / (1.f + __expf(-1.702f * v));  // quickGELU
        if (outF) outF[(size_t)row * NC + col] = v;
        if (outH) outH[(size_t)row * NC + col] = (_Float16)v;
      }
    }
  }
}

// ---------------------------------------------------------------------------
// Attention: one block per (b,h); K staged row-major, V staged TRANSPOSED in
// LDS so both score and P@V WMMA fragments gather as contiguous b128 pairs.
// ---------------------------------------------------------------------------
__global__ __launch_bounds__(256) void attn_kernel(const _Float16* __restrict__ qkv,
                                                   _Float16* __restrict__ attn_out) {
  const int bh = blockIdx.x;
  const int b = bh & 63;        // 0..63
  const int h = bh >> 6;        // 0..11
  const int tid = threadIdx.x, lane = tid & 31, wave = tid >> 5;

  __shared__ _Float16 sK[208 * 64];    // K row-major, rows 197..207 zeroed
  __shared__ _Float16 sVt[64 * 224];   // V transposed [d][k], k 197..223 zeroed
  __shared__ _Float16 sP[8][16 * 32];  // per-wave P re-layout stage

  // Stage K (row-major, zero-pad rows 197..207)
  for (int i = tid; i < 208 * 8; i += 256) {
    const int row = i >> 3, c8 = (i & 7) << 3;
    h8 zk;
#pragma unroll
    for (int j = 0; j < 8; ++j) zk[j] = (_Float16)0.f;
    if (row < LSEQ)
      zk = *(const h8*)(qkv + (size_t)(row * BTCH + b) * TDIM + DMOD +
                        h * HDIM + c8);
    *(h8*)&sK[row * 64 + c8] = zk;
  }
  // Stage V transposed: sVt[d * 224 + k] = V[k][d]
  for (int i = tid; i < LSEQ * 8; i += 256) {
    const int row = i >> 3, c8 = (i & 7) << 3;
    h8 zv = *(const h8*)(qkv + (size_t)(row * BTCH + b) * TDIM + 2 * DMOD +
                         h * HDIM + c8);
#pragma unroll
    for (int j = 0; j < 8; ++j) sVt[(c8 + j) * 224 + row] = zv[j];
  }
  // Zero-pad V columns k = 197..223
  for (int i = tid; i < 64 * 27; i += 256) {
    const int d = i / 27, k = LSEQ + (i % 27);
    sVt[d * 224 + k] = (_Float16)0.f;
  }
  __syncthreads();

  const int k0 = (lane >> 4) << 3;
  const float scale = 0.125f;  // 1/sqrt(64)

  for (int t = wave; t < 13; t += 8) {
    // ---- scores: S[j] = Q_tile(16x64) @ K_tile(16x64)^T ----
    v8f S[13];
#pragma unroll
    for (int j = 0; j < 13; ++j)
#pragma unroll
      for (int r = 0; r < 8; ++r) S[j][r] = 0.f;

    const int lq = t * 16 + (lane & 15);
#pragma unroll
    for (int kc = 0; kc < 2; ++kc) {
      v16h aq;
      if (lq < LSEQ) {
        const _Float16* p =
            qkv + (size_t)(lq * BTCH + b) * TDIM + h * HDIM + kc * 32 + k0;
        aq = pack16(*(const h8*)p, *(const h8*)(p + 16));
      } else {
        aq = zero16();
      }
#pragma unroll
      for (int j = 0; j < 13; ++j) {
        const int lk = j * 16 + (lane & 15);  // <= 207, zero-padded rows
        const _Float16* p = &sK[lk * 64 + kc * 32 + k0];
        v16h bk = pack16(*(const h8*)p, *(const h8*)(p + 16));
        S[j] = wmma32(aq, bk, S[j]);
      }
    }

    // ---- masked softmax over the 197-wide row (per-row across half-wave) ----
#pragma unroll
    for (int r = 0; r < 8; ++r) {
      float m = -1e30f;
#pragma unroll
      for (int j = 0; j < 13; ++j) {
        const int n = j * 16 + (lane & 15);
        const float v = (n < LSEQ) ? S[j][r] * scale : -1e30f;
        S[j][r] = v;
        m = fmaxf(m, v);
      }
#pragma unroll
      for (int d = 1; d < 16; d <<= 1) m = fmaxf(m, __shfl_xor(m, d, 32));
      float s = 0.f;
#pragma unroll
      for (int j = 0; j < 13; ++j) {
        const float e = __expf(S[j][r] - m);
        S[j][r] = e;
        s += e;
      }
#pragma unroll
      for (int d = 1; d < 16; d <<= 1) s += __shfl_xor(s, d, 32);
      const float inv = 1.f / s;
#pragma unroll
      for (int j = 0; j < 13; ++j) S[j][r] *= inv;
    }

    // ---- O = P @ V : re-layout P (C-frag -> A-frag) through LDS stage ----
    v8f O[4];
#pragma unroll
    for (int nt = 0; nt < 4; ++nt)
#pragma unroll
      for (int r = 0; r < 8; ++r) O[nt][r] = 0.f;

    for (int kc = 0; kc < 7; ++kc) {  // 7 chunks of 32 k-cols (224, padded)
#pragma unroll
      for (int jj = 0; jj < 2; ++jj) {
        const int j = kc * 2 + jj;
#pragma unroll
        for (int r = 0; r < 8; ++r) {
          const int m = r + ((lane >> 4) << 3);
          const int c = jj * 16 + (lane & 15);
          sP[wave][m * 32 + c] = (j < 13) ? (_Float16)S[j][r] : (_Float16)0.f;
        }
      }
      asm volatile("s_wait_dscnt 0x0" ::: "memory");  // wave-local DS ordering
      const _Float16* pp = &sP[wave][(lane & 15) * 32 + k0];
      v16h ap = pack16(*(const h8*)pp, *(const h8*)(pp + 16));
#pragma unroll
      for (int nt = 0; nt < 4; ++nt) {
        // B-frag of V from transposed stage: contiguous in k
        const _Float16* pv =
            &sVt[(nt * 16 + (lane & 15)) * 224 + kc * 32 + k0];
        v16h bv = pack16(*(const h8*)pv, *(const h8*)(pv + 16));
        O[nt] = wmma32(ap, bv, O[nt]);
      }
      asm volatile("s_wait_dscnt 0x0" ::: "memory");
    }

    // ---- store O tile (f16) into attn buffer [L,B,D] ----
#pragma unroll
    for (int nt = 0; nt < 4; ++nt) {
#pragma unroll
      for (int r = 0; r < 8; ++r) {
        const int l = t * 16 + r + ((lane >> 4) << 3);
        if (l < LSEQ)
          attn_out[(size_t)(l * BTCH + b) * DMOD + h * HDIM + nt * 16 +
                   (lane & 15)] = (_Float16)O[nt][r];
      }
    }
  }
}

// ---------------------------------------------------------------------------
// Host launcher
// ---------------------------------------------------------------------------
extern "C" void kernel_launch(void* const* d_in, const int* in_sizes, int n_in,
                              void* d_out, int out_size, void* d_ws,
                              size_t ws_size, hipStream_t stream) {
  (void)in_sizes; (void)n_in; (void)out_size; (void)ws_size;

  const float* x      = (const float*)d_in[0];
  const float* ln1_g  = (const float*)d_in[1];
  const float* ln1_b  = (const float*)d_in[2];
  const float* in_w   = (const float*)d_in[3];
  const float* in_b   = (const float*)d_in[4];
  const float* out_w  = (const float*)d_in[5];
  const float* out_b  = (const float*)d_in[6];
  const float* ln2_g  = (const float*)d_in[7];
  const float* ln2_b  = (const float*)d_in[8];
  const float* fc_w   = (const float*)d_in[9];
  const float* fc_b   = (const float*)d_in[10];
  const float* pj_w   = (const float*)d_in[11];
  const float* pj_b   = (const float*)d_in[12];
  const float* eh_w   = (const float*)d_in[13];
  const float* eh_b   = (const float*)d_in[14];
  const float* et_w   = (const float*)d_in[15];
  const float* et_b   = (const float*)d_in[16];
  const float* r1_w   = (const float*)d_in[17];
  const float* r1_b   = (const float*)d_in[18];
  const float* r2_w   = (const float*)d_in[19];
  const float* r2_b   = (const float*)d_in[20];

  char* base = (char*)d_ws;
  size_t off = 0;
  auto alloc = [&](size_t bytes) -> void* {
    void* p = base + off;
    off = (off + bytes + 255) & ~(size_t)255;
    return p;
  };

  _Float16* w_in_h = (_Float16*)alloc((size_t)TDIM * DMOD * 2);
  _Float16* w_out_h = (_Float16*)alloc((size_t)DMOD * DMOD * 2);
  _Float16* w_fc_h = (_Float16*)alloc((size_t)DFF * DMOD * 2);
  _Float16* w_eh_h = (_Float16*)alloc((size_t)4 * DFF * DMOD * 2);
  _Float16* w_pj_h = (_Float16*)alloc((size_t)DMOD * DFF * 2);
  _Float16* w_et_h = (_Float16*)alloc((size_t)4 * DMOD * DFF * 2);
  _Float16* lnx_h = (_Float16*)alloc((size_t)NTOK * DMOD * 2);    // ln1 then ln2
  _Float16* big_h = (_Float16*)alloc((size_t)NTOK * DFF * 2);     // qkv then head
  _Float16* attn_h = (_Float16*)alloc((size_t)NTOK * DMOD * 2);
  float* x1 = (float*)alloc((size_t)NTOK * DMOD * 4);
  float* r1 = (float*)alloc((size_t)NTOK * 5 * 4);
  float* r2 = (float*)alloc((size_t)NTOK * 5 * 4);

  auto cvt = [&](const float* s, _Float16* d, size_t n) {
    unsigned blocks = (unsigned)((n + 255) / 256);
    cvt_kernel<<<blocks, 256, 0, stream>>>(s, d, n);
  };
  cvt(in_w, w_in_h, (size_t)TDIM * DMOD);
  cvt(out_w, w_out_h, (size_t)DMOD * DMOD);
  cvt(fc_w, w_fc_h, (size_t)DFF * DMOD);
  cvt(eh_w, w_eh_h, (size_t)4 * DFF * DMOD);
  cvt(pj_w, w_pj_h, (size_t)DMOD * DFF);
  cvt(et_w, w_et_h, (size_t)4 * DMOD * DFF);

  const unsigned rowBlocks = NTOK / 8;  // 1576

  // 1) ln1(x) -> f16
  ln_kernel<<<rowBlocks, 256, 0, stream>>>(x, ln1_g, ln1_b, lnx_h, NTOK);

  // 2) qkv = ln1x @ in_proj^T + b  -> f16 [NTOK, 2304]
  gemm_wmma<1><<<dim3(NTOK / 64, TDIM / 128), 256, 0, stream>>>(
      lnx_h, DMOD, TDIM, w_in_h, nullptr, in_b, nullptr, nullptr, nullptr,
      nullptr, big_h, 0);

  // 3) attention -> f16 [NTOK, 768]
  attn_kernel<<<BTCH * NHEAD, 256, 0, stream>>>(big_h, attn_h);

  // 4) x1 = x + attn @ out_proj^T + b  -> f32
  gemm_wmma<1><<<dim3(NTOK / 64, DMOD / 128), 256, 0, stream>>>(
      attn_h, DMOD, DMOD, w_out_h, nullptr, out_b, nullptr, nullptr, x, x1,
      nullptr, 0);

  // 5) ln2(x1) -> f16
  ln_kernel<<<rowBlocks, 256, 0, stream>>>(x1, ln2_g, ln2_b, lnx_h, NTOK);

  // 6) r1 = softmax(ln2x @ r1_w^T + r1_b)
  router_kernel<<<rowBlocks, 256, 0, stream>>>(lnx_h, DMOD, r1_w, r1_b, r1, NTOK);

  // 7) out_head = quickGELU( sum_e r1_e * (ln2x @ Wh_e^T + bh_e) ) -> f16
  gemm_wmma<5><<<dim3(NTOK / 64, DFF / 128), 256, 0, stream>>>(
      lnx_h, DMOD, DFF, w_fc_h, w_eh_h, fc_b, eh_b, r1, nullptr, nullptr,
      big_h, 1);

  // 8) r2 = softmax(out_head @ r2_w^T + r2_b)
  router_kernel<<<rowBlocks, 256, 0, stream>>>(big_h, DFF, r2_w, r2_b, r2, NTOK);

  // 9) out = x1 + sum_e r2_e * (out_head @ Wt_e^T + bt_e) -> f32 d_out
  gemm_wmma<5><<<dim3(NTOK / 64, DMOD / 128), 256, 0, stream>>>(
      big_h, DFF, DMOD, w_pj_h, w_et_h, pj_b, et_b, r2, x1, (float*)d_out,
      nullptr, 0);
}
